// Attention_19662360281663
// MI455X (gfx1250) — matrix operand order repaired and checked by reading, not compiled, not run on hardware
//
#include <hip/hip_runtime.h>
#include <hip/hip_bf16.h>

#define B_  4
#define S_  1024
#define E_  256
#define H_  12
#define EH_ (E_ * H_)   // 3072
#define M_  (B_ * S_)   // 4096

typedef __attribute__((ext_vector_type(16))) _Float16 v16h;
typedef __attribute__((ext_vector_type(8)))  _Float16 v8h;
typedef __attribute__((ext_vector_type(8)))  float    v8f;

static __device__ __forceinline__ v8f wmma_f16(v16h a, v16h b, v8f c) {
    return __builtin_amdgcn_wmma_f32_16x16x32_f16(
        /*neg_a=*/false, a, /*neg_b=*/false, b,
        /*c_mod=*/(short)0, c, /*reuse_a=*/false, /*reuse_b=*/false);
}

// A-matrix fragment, 16x32 f16 (M x K), source row-major with leading dim `ld` (halves).
// ISA 7.12.2: lanes 0-15 -> M=lane, K={0..7,16..23}; lanes 16-31 -> M=lane-16, K={8..15,24..31}
static __device__ __forceinline__ v16h load_frag_a(const _Float16* p, int ld) {
    const int lane = threadIdx.x & 31;
    const int r    = lane & 15;
    const int hi   = (lane >> 4) & 1;
    const _Float16* row = p + r * ld + hi * 8;
    v8h lo = *(const v8h*)(row);        // K = hi*8 .. hi*8+7
    v8h hv = *(const v8h*)(row + 16);   // K = 16+hi*8 .. 16+hi*8+7
    v16h out;
#pragma unroll
    for (int i = 0; i < 8; ++i) { out[i] = lo[i]; out[i + 8] = hv[i]; }
    return out;
}

// B-matrix fragment, 32x16 f16 (K x N), loaded from TRANSPOSED storage (N x K row-major).
// lanes 0-15 -> N=lane, K={0..7,16..23}; lanes 16-31 -> N=lane-16, K={8..15,24..31}
static __device__ __forceinline__ v16h load_frag_b(const _Float16* p, int ld) {
    const int lane = threadIdx.x & 31;
    const int c    = lane & 15;
    const int hi   = (lane >> 4) & 1;
    const _Float16* row = p + c * ld + hi * 8;
    v8h lo = *(const v8h*)(row);
    v8h hv = *(const v8h*)(row + 16);
    v16h out;
#pragma unroll
    for (int i = 0; i < 8; ++i) { out[i] = lo[i]; out[i + 8] = hv[i]; }
    return out;
}

// ---------------- conversion kernels ----------------

__global__ void cvt_f32_to_f16_kernel(const float* __restrict__ src,
                                      _Float16* __restrict__ dst, int n) {
    int i = blockIdx.x * blockDim.x + threadIdx.x;
    if (i < n) dst[i] = (_Float16)src[i];
}

// src: (K, N) f32 row-major -> dst: (N, K) f16 row-major
__global__ void transpose_cvt_kernel(const float* __restrict__ src,
                                     _Float16* __restrict__ dst, int K, int N) {
    int i = blockIdx.x * blockDim.x + threadIdx.x;
    if (i < K * N) {
        int n = i / K, k = i % K;
        dst[i] = (_Float16)src[k * N + n];
    }
}

// ---------------- fused QKV projection ----------------
__global__ __launch_bounds__(256) void qkv_gemm_kernel(
    const _Float16* __restrict__ x16,
    const _Float16* __restrict__ wqT, const _Float16* __restrict__ wkT,
    const _Float16* __restrict__ wvT,
    const float* __restrict__ bq, const float* __restrict__ bk,
    const float* __restrict__ bv,
    _Float16* __restrict__ q16, _Float16* __restrict__ k16,
    _Float16* __restrict__ vT16)
{
    const int wave = blockIdx.x * (blockDim.x >> 5) + (threadIdx.x >> 5);
    const int NT = (3 * EH_) / 64;               // 144 n-tiles total
    const int mt = wave / NT;
    const int nt = wave % NT;
    if (mt >= M_ / 16) return;
    const int which = nt / (EH_ / 64);           // 0=q 1=k 2=v
    const int n0 = (nt % (EH_ / 64)) * 64;
    const _Float16* wT  = (which == 0) ? wqT : (which == 1) ? wkT : wvT;
    const float*    bia = (which == 0) ? bq  : (which == 1) ? bk  : bv;

    v8f acc[4];
#pragma unroll
    for (int t = 0; t < 4; ++t) acc[t] = v8f{};

#pragma unroll
    for (int k0 = 0; k0 < E_; k0 += 32) {
        v16h a = load_frag_a(x16 + mt * 16 * E_ + k0, E_);
#pragma unroll
        for (int t = 0; t < 4; ++t) {
            v16h b = load_frag_b(wT + (n0 + t * 16) * E_ + k0, E_);
            acc[t] = wmma_f16(a, b, acc[t]);
        }
    }

    const int lane = threadIdx.x & 31;
    const int cn   = lane & 15;
    const int hi8  = ((lane >> 4) & 1) * 8;
#pragma unroll
    for (int t = 0; t < 4; ++t) {
        const int n = n0 + t * 16 + cn;
        const int h = n >> 8, e = n & 255;
        const float bb = bia[n];
#pragma unroll
        for (int j = 0; j < 8; ++j) {
            const int m = mt * 16 + j + hi8;
            const int b = m >> 10, s = m & (S_ - 1);
            const float val = acc[t][j] + bb;
            if (which == 2)
                vT16[((size_t)((b * H_ + h) * E_ + e)) * S_ + s] = (_Float16)val;
            else if (which == 0)
                q16[((size_t)((b * H_ + h) * S_ + s)) * E_ + e] = (_Float16)val;
            else
                k16[((size_t)((b * H_ + h) * S_ + s)) * E_ + e] = (_Float16)val;
        }
    }
}

// ---------------- flash attention with async K/V staging ----------------
// Block = 8 waves, all same (b,h), consecutive q-blocks -> K/V tiles shared.
// K/V tiles are DMAed into double-buffered LDS with GLOBAL_LOAD_ASYNC_TO_LDS_B128
// (ASYNCcnt), overlapping the next tile's copy with the current tile's WMMAs.
__global__ __launch_bounds__(256) void attn_kernel(
    const _Float16* __restrict__ q16, const _Float16* __restrict__ k16,
    const _Float16* __restrict__ vT16, _Float16* __restrict__ mrg16)
{
    __shared__ __align__(16) _Float16 kbuf[2][32 * E_];   // 2 x 16 KB (32 keys x 256)
    __shared__ __align__(16) _Float16 vbuf[2][E_ * 32];   // 2 x 16 KB (256 e x 32 keys)
    __shared__ __align__(16) _Float16 lds_p[8][16 * 32];  // per-wave probs tile (8 KB)

    const int tid  = threadIdx.x;
    const int wid  = tid >> 5;
    const int lane = tid & 31;
    const int cn   = lane & 15;
    const int hi8  = ((lane >> 4) & 1) * 8;

    const int bh    = blockIdx.x >> 3;        // 0..47 = b*H + h
    const int qbase = (blockIdx.x & 7) * 8;   // first q-block of this workgroup
    const int qb    = qbase + wid;            // this wave's q-block

    const _Float16* qp = q16  + (size_t)bh * S_ * E_ + (size_t)qb * 16 * E_;
    const _Float16* kp = k16  + (size_t)bh * S_ * E_;
    const _Float16* vp = vT16 + (size_t)bh * E_ * S_;

    // cache Q fragments (16 x 256 -> 8 chunks of K=32)
    v16h qfrag[8];
#pragma unroll
    for (int k0 = 0; k0 < E_; k0 += 32) qfrag[k0 / 32] = load_frag_a(qp + k0, E_);

    v8f ctx[16];
#pragma unroll
    for (int t = 0; t < 16; ++t) ctx[t] = v8f{};
    float mrow[8], lrow[8];
#pragma unroll
    for (int j = 0; j < 8; ++j) { mrow[j] = -3.0e38f; lrow[j] = 0.f; }

    const int nkb     = (qb + 2) >> 1;            // this wave's causal bound
    const int nkb_max = (qbase + 9) >> 1;         // uniform block trip count

    // cooperative async copy of key-block kb into LDS buffer bufi
    auto issue_loads = [&](int kb, int bufi) {
        // K tile: contiguous 8192 halves (32 rows x 256)
        const _Float16* gk = kp + (size_t)kb * 32 * E_;
        _Float16* lk = &kbuf[bufi][0];
#pragma unroll
        for (int p = 0; p < 4; ++p) {
            const int off = (p * 256 + tid) * 8;           // halves, 16B chunks
            unsigned laddr = (unsigned)(uintptr_t)(lk + off);
            const _Float16* ga = gk + off;
            asm volatile("global_load_async_to_lds_b128 %0, %1, off"
                         :: "v"(laddr), "v"(ga) : "memory");
        }
        // V tile: 256 rows x 32 halves, global row stride S_
        _Float16* lv = &vbuf[bufi][0];
#pragma unroll
        for (int p = 0; p < 4; ++p) {
            const int t   = p * 256 + tid;
            const int row = t >> 2;
            const int ch  = (t & 3) * 8;
            unsigned laddr = (unsigned)(uintptr_t)(lv + row * 32 + ch);
            const _Float16* ga = vp + (size_t)row * S_ + kb * 32 + ch;
            asm volatile("global_load_async_to_lds_b128 %0, %1, off"
                         :: "v"(laddr), "v"(ga) : "memory");
        }
    };

    issue_loads(0, 0);

    for (int kb = 0; kb < nkb_max; ++kb) {
        asm volatile("s_wait_asynccnt 0x0" ::: "memory");  // own DMAs done
        __syncthreads();                                   // everyone's DMAs visible
        if (kb + 1 < nkb_max) issue_loads(kb + 1, (kb + 1) & 1);

        if (kb < nkb) {
            const _Float16* lk = &kbuf[kb & 1][0];
            const _Float16* lv = &vbuf[kb & 1][0];

            v8f s0 = v8f{}, s1 = v8f{};
#pragma unroll
            for (int kc = 0; kc < 8; ++kc) {
                const int k0 = kc * 32;
                v16h b0 = load_frag_b(lk + k0, E_);            // keys 0..15 of tile
                v16h b1 = load_frag_b(lk + 16 * E_ + k0, E_);  // keys 16..31
                s0 = wmma_f16(qfrag[kc], b0, s0);
                s1 = wmma_f16(qfrag[kc], b1, s1);
            }
            // scale (1/sqrt(256)) + causal mask (== -10000 additive mask after expf)
            const int col0 = kb * 32 + cn;
            const int col1 = col0 + 16;
            float alpha[8];
#pragma unroll
            for (int j = 0; j < 8; ++j) {
                const int row = qb * 16 + j + hi8;
                float t0 = s0[j] * 0.0625f + ((col0 > row) ? -10000.0f : 0.0f);
                float t1 = s1[j] * 0.0625f + ((col1 > row) ? -10000.0f : 0.0f);
                float mx = fmaxf(t0, t1);
#pragma unroll
                for (int d = 1; d < 16; d <<= 1) mx = fmaxf(mx, __shfl_xor(mx, d, 32));
                const float mnew = fmaxf(mrow[j], mx);
                alpha[j] = __expf(mrow[j] - mnew);
                const float p0 = __expf(t0 - mnew);
                const float p1 = __expf(t1 - mnew);
                float rs = p0 + p1;
#pragma unroll
                for (int d = 1; d < 16; d <<= 1) rs += __shfl_xor(rs, d, 32);
                lrow[j] = lrow[j] * alpha[j] + rs;
                mrow[j] = mnew;
                s0[j] = p0; s1[j] = p1;
            }
#pragma unroll
            for (int t = 0; t < 16; ++t)
#pragma unroll
                for (int j = 0; j < 8; ++j) ctx[t][j] *= alpha[j];

            // probs C-layout -> A-layout via per-wave LDS tile
            _Float16* pt = &lds_p[wid][0];
#pragma unroll
            for (int j = 0; j < 8; ++j) {
                const int row = j + hi8;
                pt[row * 32 + cn]      = (_Float16)s0[j];
                pt[row * 32 + 16 + cn] = (_Float16)s1[j];
            }
            asm volatile("s_wait_dscnt 0x0" ::: "memory");
            v16h pa = load_frag_a(pt, 32);

            // ctx += P @ V  (V tile in LDS as (256 e) x (32 kk))
#pragma unroll
            for (int t = 0; t < 16; ++t) {
                v16h b = load_frag_b(lv + t * 16 * 32, 32);
                ctx[t] = wmma_f16(pa, b, ctx[t]);
            }
        }
    }

    // normalize and write merged (B, S, H*E) f16
    float inv_l[8];
#pragma unroll
    for (int j = 0; j < 8; ++j) inv_l[j] = 1.0f / lrow[j];
    const int b = bh / H_, h = bh % H_;
#pragma unroll
    for (int t = 0; t < 16; ++t) {
        const int e = t * 16 + cn;
#pragma unroll
        for (int j = 0; j < 8; ++j) {
            const int srow = qb * 16 + j + hi8;
            mrg16[((size_t)(b * S_ + srow)) * EH_ + h * E_ + e] =
                (_Float16)(ctx[t][j] * inv_l[j]);
        }
    }
}

// ---------------- final FC: (4096,3072) @ (3072,256) + bias -> f32 out ----------------
__global__ __launch_bounds__(256) void fc_gemm_kernel(
    const _Float16* __restrict__ a16, const _Float16* __restrict__ wT,
    const float* __restrict__ bias, float* __restrict__ out)
{
    const int wave = blockIdx.x * (blockDim.x >> 5) + (threadIdx.x >> 5);
    const int NT = E_ / 64;              // 4
    const int mt = wave / NT;
    const int nt = wave % NT;
    if (mt >= M_ / 16) return;
    const int n0 = nt * 64;

    v8f acc[4];
#pragma unroll
    for (int t = 0; t < 4; ++t) acc[t] = v8f{};

    for (int k0 = 0; k0 < EH_; k0 += 32) {
        v16h a = load_frag_a(a16 + (size_t)mt * 16 * EH_ + k0, EH_);
#pragma unroll
        for (int t = 0; t < 4; ++t) {
            v16h b = load_frag_b(wT + (size_t)(n0 + t * 16) * EH_ + k0, EH_);
            acc[t] = wmma_f16(a, b, acc[t]);
        }
    }

    const int lane = threadIdx.x & 31;
    const int cn   = lane & 15;
    const int hi8  = ((lane >> 4) & 1) * 8;
#pragma unroll
    for (int t = 0; t < 4; ++t) {
        const int n = n0 + t * 16 + cn;
        const float bb = bias[n];
#pragma unroll
        for (int j = 0; j < 8; ++j) {
            const int m = mt * 16 + j + hi8;
            out[(size_t)m * E_ + n] = acc[t][j] + bb;
        }
    }
}

extern "C" void kernel_launch(void* const* d_in, const int* in_sizes, int n_in,
                              void* d_out, int out_size, void* d_ws, size_t ws_size,
                              hipStream_t stream) {
    const float* x   = (const float*)d_in[0];
    // d_in[1] = attention_mask (causal, reproduced inline in attn_kernel)
    const float* Wq  = (const float*)d_in[2];
    const float* bq  = (const float*)d_in[3];
    const float* Wk  = (const float*)d_in[4];
    const float* bk  = (const float*)d_in[5];
    const float* Wv  = (const float*)d_in[6];
    const float* bv  = (const float*)d_in[7];
    const float* Wfc = (const float*)d_in[8];
    const float* bfc = (const float*)d_in[9];
    float* out = (float*)d_out;

    // workspace carve-up (f16 elements)
    _Float16* p    = (_Float16*)d_ws;
    _Float16* x16  = p; p += (size_t)M_ * E_;        // 1,048,576
    _Float16* wqT  = p; p += (size_t)EH_ * E_;       //   786,432
    _Float16* wkT  = p; p += (size_t)EH_ * E_;
    _Float16* wvT  = p; p += (size_t)EH_ * E_;
    _Float16* wfcT = p; p += (size_t)EH_ * E_;
    _Float16* q16  = p; p += (size_t)B_ * H_ * S_ * E_;   // 12,582,912
    _Float16* k16  = p; p += (size_t)B_ * H_ * S_ * E_;
    _Float16* vT16 = p; p += (size_t)B_ * H_ * S_ * E_;
    _Float16* mg16 = p; p += (size_t)M_ * EH_;            // 12,582,912

    const int nx = M_ * E_;
    cvt_f32_to_f16_kernel<<<(nx + 255) / 256, 256, 0, stream>>>(x, x16, nx);
    const int nw = EH_ * E_;
    transpose_cvt_kernel<<<(nw + 255) / 256, 256, 0, stream>>>(Wq,  wqT,  E_,  EH_);
    transpose_cvt_kernel<<<(nw + 255) / 256, 256, 0, stream>>>(Wk,  wkT,  E_,  EH_);
    transpose_cvt_kernel<<<(nw + 255) / 256, 256, 0, stream>>>(Wv,  wvT,  E_,  EH_);
    transpose_cvt_kernel<<<(nw + 255) / 256, 256, 0, stream>>>(Wfc, wfcT, EH_, E_);

    // QKV: 256 m-tiles x 144 n-tiles = 36864 wave-tiles / 8 waves per block
    qkv_gemm_kernel<<<4608, 256, 0, stream>>>(x16, wqT, wkT, wvT, bq, bk, bv,
                                              q16, k16, vT16);
    // attention: 48 (b,h) x 8 q-groups = 384 blocks of 8 waves
    attn_kernel<<<384, 256, 0, stream>>>(q16, k16, vT16, mg16);
    // FC: 256 m-tiles x 4 n-tiles = 1024 wave-tiles / 8 per block
    fc_gemm_kernel<<<128, 256, 0, stream>>>(mg16, wfcT, bfc, out);
}
